// Nonlocal_31636729102404
// MI455X (gfx1250) — matrix operand order repaired and checked by reading, not compiled, not run on hardware
//
#include <hip/hip_runtime.h>
#include <cmath>

#define NB  8
#define CH  256
#define HWD 4096

typedef unsigned short u16;
typedef __attribute__((ext_vector_type(16))) __bf16 v16bf;
typedef __attribute__((ext_vector_type(8)))  float  v8f;

__device__ __forceinline__ u16 f2bf(float f) {
    union { float f; unsigned u; } v; v.f = f;
    unsigned u = v.u;
    u += 0x7FFFu + ((u >> 16) & 1u);        // round-to-nearest-even
    return (u16)(u >> 16);
}

// A fragment (16-bit, 16x32): elements 0..7 at p, 8..15 at p+16 elements
__device__ __forceinline__ v16bf load_frag_a(const u16* p) {
    union { uint4 q[2]; v16bf v; } u;
    u.q[0] = *reinterpret_cast<const uint4*>(p);
    u.q[1] = *reinterpret_cast<const uint4*>(p + 16);
    return u.v;
}
// B fragment (16-bit, 32x16): elements 0..7 at p, 8..15 at p+16 elements
__device__ __forceinline__ v16bf load_frag_b(const u16* p) {
    union { uint4 q[2]; v16bf v; } u;
    u.q[0] = *reinterpret_cast<const uint4*>(p);
    u.q[1] = *reinterpret_cast<const uint4*>(p + 16);
    return u.v;
}
__device__ __forceinline__ v8f wmma_bf16(v16bf a, v16bf b, v8f c) {
    return __builtin_amdgcn_wmma_f32_16x16x32_bf16(false, a, false, b, (short)0, c, false, false);
}
__device__ __forceinline__ uint4 pack8(const float* f) {
    union { u16 s[8]; uint4 q; } u;
#pragma unroll
    for (int i = 0; i < 8; ++i) u.s[i] = f2bf(f[i]);
    return u.q;
}
__device__ __forceinline__ float redmax16(float v) {
    v = fmaxf(v, __shfl_xor(v, 1, 32));
    v = fmaxf(v, __shfl_xor(v, 2, 32));
    v = fmaxf(v, __shfl_xor(v, 4, 32));
    v = fmaxf(v, __shfl_xor(v, 8, 32));
    return v;
}
__device__ __forceinline__ float redsum16(float v) {
    v += __shfl_xor(v, 1, 32);
    v += __shfl_xor(v, 2, 32);
    v += __shfl_xor(v, 4, 32);
    v += __shfl_xor(v, 8, 32);
    return v;
}

// ---------------- weight fp32 -> bf16 ----------------
__global__ __launch_bounds__(256) void convert_w_kernel(
    const float* wt, const float* wp, const float* wg, const float* wo,
    u16* bq, u16* bk, u16* bv, u16* bo)
{
    int i = blockIdx.x * 256 + threadIdx.x;   // 65536 elems each
    bq[i] = f2bf(wt[i]);
    bk[i] = f2bf(wp[i]);
    bv[i] = f2bf(wg[i]);
    bo[i] = f2bf(wo[i]);
}

// ---------------- x [N,C,HW] fp32 -> xT [N,HW,C] bf16 (LDS transpose) -------
__global__ __launch_bounds__(256) void convert_x_kernel(const float* x, u16* xT)
{
    __shared__ float tile[32][33];
    int n  = blockIdx.z;
    int c0 = blockIdx.y * 32;
    int p0 = blockIdx.x * 32;
#pragma unroll
    for (int j = 0; j < 4; ++j) {
        int c = c0 + threadIdx.y + j * 8;
        tile[threadIdx.y + j * 8][threadIdx.x] =
            x[((size_t)(n * CH + c)) * HWD + p0 + threadIdx.x];
    }
    __syncthreads();
#pragma unroll
    for (int j = 0; j < 4; ++j) {
        int pix = p0 + threadIdx.y + j * 8;
        xT[((size_t)(n * HWD + pix)) * CH + c0 + threadIdx.x] =
            f2bf(tile[threadIdx.x][threadIdx.y + j * 8]);
    }
}

// ---------------- Q,K GEMM: D[o,pix] = W · xT^T ; store [N,HW,C] ------------
__global__ __launch_bounds__(128) void qk_kernel(
    const u16* xT, const u16* wq, const u16* wk,
    const float* bth, const float* bph, u16* Q, u16* K)
{
    int wave = threadIdx.x >> 5, lane = threadIdx.x & 31;
    int hi = lane >> 4, ln = lane & 15;
    int n  = blockIdx.x >> 8;
    int p0 = (blockIdx.x & 255) * 16;
    const u16* xrow = xT + ((size_t)(n * HWD + p0 + ln)) * CH;

    v8f dq[4], dk[4];
#pragma unroll
    for (int t = 0; t < 4; ++t) { dq[t] = (v8f)(0.f); dk[t] = (v8f)(0.f); }

#pragma unroll
    for (int kc = 0; kc < 8; ++kc) {
        v16bf xb = load_frag_b(xrow + kc * 32 + hi * 8);
#pragma unroll
        for (int t = 0; t < 4; ++t) {
            int ot = wave * 4 + t;
            const u16* wrow = wq + (size_t)(ot * 16 + ln) * CH + kc * 32 + hi * 8;
            dq[t] = wmma_bf16(load_frag_a(wrow), xb, dq[t]);
            const u16* krow = wk + (size_t)(ot * 16 + ln) * CH + kc * 32 + hi * 8;
            dk[t] = wmma_bf16(load_frag_a(krow), xb, dk[t]);
        }
    }
#pragma unroll
    for (int t = 0; t < 4; ++t) {
        int ot = wave * 4 + t;
        float vq[8], vk[8];
#pragma unroll
        for (int v = 0; v < 8; ++v) {
            int o = ot * 16 + 8 * hi + v;
            vq[v] = dq[t][v] + bth[o];
            vk[v] = dk[t][v] + bph[o];
        }
        size_t base = ((size_t)(n * HWD + p0 + ln)) * CH + ot * 16 + 8 * hi;
        *reinterpret_cast<uint4*>(Q + base) = pack8(vq);
        *reinterpret_cast<uint4*>(K + base) = pack8(vk);
    }
}

// ---------------- V GEMM: D[pix,o] = xT · W^T ; store Vt [N,C,HW] -----------
__global__ __launch_bounds__(128) void v_kernel(
    const u16* xT, const u16* wv, const float* bg, u16* Vt)
{
    int wave = threadIdx.x >> 5, lane = threadIdx.x & 31;
    int hi = lane >> 4, ln = lane & 15;
    int n  = blockIdx.x >> 8;
    int p0 = (blockIdx.x & 255) * 16;
    const u16* xrow = xT + ((size_t)(n * HWD + p0 + ln)) * CH;

    v8f d[4];
#pragma unroll
    for (int t = 0; t < 4; ++t) d[t] = (v8f)(0.f);

#pragma unroll
    for (int kc = 0; kc < 8; ++kc) {
        v16bf xa = load_frag_a(xrow + kc * 32 + hi * 8);
#pragma unroll
        for (int t = 0; t < 4; ++t) {
            int ot = wave * 4 + t;
            const u16* wrow = wv + (size_t)(ot * 16 + ln) * CH + kc * 32 + hi * 8;
            d[t] = wmma_bf16(xa, load_frag_b(wrow), d[t]);
        }
    }
#pragma unroll
    for (int t = 0; t < 4; ++t) {
        int ot = wave * 4 + t;
        int o  = ot * 16 + ln;
        float bb = bg[o];
        float vv[8];
#pragma unroll
        for (int v = 0; v < 8; ++v) vv[v] = d[t][v] + bb;
        size_t base = ((size_t)(n * CH + o)) * HWD + p0 + 8 * hi;
        *reinterpret_cast<uint4*>(Vt + base) = pack8(vv);
    }
}

// ---------------- flash attention: O^T[c,h] = V · P^T ; store [N,HW,C] ------
// 64-wide K/V chunks: halves flash-rescale VALU, barriers and stat bookkeeping
// per WMMA vs a 32-wide chunk. Q fragments are re-read from WGP$/L2 each
// iteration instead of register-cached to stay under the VGPR budget.
__global__ __launch_bounds__(128) void attn_kernel(
    const u16* Q, const u16* K, const u16* Vt, u16* TPG)
{
    int wave = threadIdx.x >> 5, lane = threadIdx.x & 31;
    int hi = lane >> 4, ln = lane & 15;
    int n  = blockIdx.x;
    int h0 = (blockIdx.y * 4 + wave) * 16;
    const float scale = 0.0625f; // C^-0.5

    const u16* Qrow = Q + ((size_t)(n * HWD + h0 + ln)) * CH;
    const u16* Kn   = K + (size_t)n * HWD * CH;
    const u16* Vn   = Vt + (size_t)n * CH * HWD;

    __shared__ __align__(16) u16 Pb[4][16][64];
    __shared__ float Fb[4][16];

    v8f acc[16];
#pragma unroll
    for (int t = 0; t < 16; ++t) acc[t] = (v8f)(0.f);
    float mrow[8], lrow[8];
#pragma unroll
    for (int v = 0; v < 8; ++v) { mrow[v] = -1e30f; lrow[v] = 0.f; }

    for (int p0 = 0; p0 < HWD; p0 += 64) {
        // ---- S = Q K^T for 16 rows x 64 cols ----
        v8f s[4];
#pragma unroll
        for (int j = 0; j < 4; ++j) s[j] = (v8f)(0.f);
#pragma unroll
        for (int kc = 0; kc < 8; ++kc) {
            v16bf qf = load_frag_a(Qrow + kc * 32 + hi * 8);
#pragma unroll
            for (int j = 0; j < 4; ++j) {
                v16bf kf = load_frag_b(Kn + (size_t)(p0 + j * 16 + ln) * CH + kc * 32 + hi * 8);
                s[j] = wmma_bf16(qf, kf, s[j]);
            }
        }
        // warm the L2->WGP path for the next chunk's K rows
        if (p0 + 64 < HWD) {
            __builtin_prefetch(Kn + (size_t)(p0 + 64 + lane) * CH, 0, 1);
            __builtin_prefetch(Kn + (size_t)(p0 + 96 + lane) * CH, 0, 1);
        }
        // ---- online softmax over this 64-wide chunk (rows = v + 8*hi) ----
        float fac[8];
#pragma unroll
        for (int v = 0; v < 8; ++v) {
            float a0 = s[0][v] * scale, a1 = s[1][v] * scale;
            float a2 = s[2][v] * scale, a3 = s[3][v] * scale;
            float cm = redmax16(fmaxf(fmaxf(a0, a1), fmaxf(a2, a3)));
            float mnew = fmaxf(mrow[v], cm);
            fac[v] = __expf(mrow[v] - mnew);
            float e0 = __expf(a0 - mnew);
            float e1 = __expf(a1 - mnew);
            float e2 = __expf(a2 - mnew);
            float e3 = __expf(a3 - mnew);
            float rs = redsum16(e0 + e1 + e2 + e3);
            lrow[v] = lrow[v] * fac[v] + rs;
            mrow[v] = mnew;
            int r = v + 8 * hi;
            Pb[wave][r][ln]      = f2bf(e0);
            Pb[wave][r][16 + ln] = f2bf(e1);
            Pb[wave][r][32 + ln] = f2bf(e2);
            Pb[wave][r][48 + ln] = f2bf(e3);
        }
        if (ln == 0) {
#pragma unroll
            for (int v = 0; v < 8; ++v) Fb[wave][8 * hi + v] = fac[v];
        }
        __syncthreads();
        float f = Fb[wave][ln];
        v16bf pf0 = load_frag_b(&Pb[wave][ln][hi * 8]);
        v16bf pf1 = load_frag_b(&Pb[wave][ln][32 + hi * 8]);
        __syncthreads();
#pragma unroll
        for (int t = 0; t < 16; ++t) {
#pragma unroll
            for (int v = 0; v < 8; ++v) acc[t][v] *= f;
        }
        // ---- O^T[c,h] += V[c,p] * P^T[p,h] ----
#pragma unroll
        for (int ct = 0; ct < 16; ++ct) {
            const u16* vrow = Vn + (size_t)(ct * 16 + ln) * HWD + p0;
            acc[ct] = wmma_bf16(load_frag_a(vrow + hi * 8),      pf0, acc[ct]);
            acc[ct] = wmma_bf16(load_frag_a(vrow + 32 + hi * 8), pf1, acc[ct]);
            if (p0 + 64 < HWD) __builtin_prefetch(vrow + 64, 0, 1);
        }
    }

    if (ln == 0) {
#pragma unroll
        for (int v = 0; v < 8; ++v) Fb[wave][8 * hi + v] = 1.0f / lrow[v];
    }
    __syncthreads();
    float inv = Fb[wave][ln];
#pragma unroll
    for (int ct = 0; ct < 16; ++ct) {
        float vv[8];
#pragma unroll
        for (int v = 0; v < 8; ++v) vv[v] = acc[ct][v] * inv;
        size_t base = ((size_t)(n * HWD + h0 + ln)) * CH + ct * 16 + 8 * hi;
        *reinterpret_cast<uint4*>(TPG + base) = pack8(vv);
    }
}

// ---------------- output conv: P[o,pix] fp32 [N,C,HW] -----------------------
__global__ __launch_bounds__(128) void pconv_kernel(
    const u16* TPG, const u16* wo, const float* bo, float* P)
{
    int wave = threadIdx.x >> 5, lane = threadIdx.x & 31;
    int hi = lane >> 4, ln = lane & 15;
    int n  = blockIdx.x >> 8;
    int p0 = (blockIdx.x & 255) * 16;
    const u16* trow = TPG + ((size_t)(n * HWD + p0 + ln)) * CH;

    v8f d[4];
#pragma unroll
    for (int t = 0; t < 4; ++t) d[t] = (v8f)(0.f);

#pragma unroll
    for (int kc = 0; kc < 8; ++kc) {
        v16bf tb = load_frag_b(trow + kc * 32 + hi * 8);
#pragma unroll
        for (int t = 0; t < 4; ++t) {
            int ot = wave * 4 + t;
            const u16* wrow = wo + (size_t)(ot * 16 + ln) * CH + kc * 32 + hi * 8;
            d[t] = wmma_bf16(load_frag_a(wrow), tb, d[t]);
        }
    }
#pragma unroll
    for (int t = 0; t < 4; ++t) {
        int ot = wave * 4 + t;
#pragma unroll
        for (int v = 0; v < 8; ++v) {
            int o = ot * 16 + 8 * hi + v;
            P[((size_t)(n * CH + o)) * HWD + p0 + ln] = d[t][v] + bo[o];
        }
    }
}

// ---------------- BN stats: one block per channel ---------------------------
__global__ __launch_bounds__(256) void bn_stats_kernel(
    const float* P, float* meanArr, float* varArr)
{
    __shared__ float ss[256], s2[256];
    int c = blockIdx.x, t = threadIdx.x;
    float s = 0.f, q = 0.f;
    for (int i = t; i < NB * HWD; i += 256) {
        int n = i >> 12, pix = i & (HWD - 1);
        float v = P[((size_t)(n * CH + c)) * HWD + pix];
        s += v; q += v * v;
    }
    ss[t] = s; s2[t] = q;
    __syncthreads();
    for (int st = 128; st > 0; st >>= 1) {
        if (t < st) { ss[t] += ss[t + st]; s2[t] += s2[t + st]; }
        __syncthreads();
    }
    if (t == 0) {
        float m = ss[0] * (1.0f / (NB * HWD));
        meanArr[c] = m;
        varArr[c]  = s2[0] * (1.0f / (NB * HWD)) - m * m;
    }
}

// ---------------- BN + gate elementwise -------------------------------------
__global__ __launch_bounds__(256) void bn_final_kernel(
    const float* P, const float* x, const float* gamma, const float* beta,
    const float* wgate, const float* meanArr, const float* varArr, float* out)
{
    size_t idx = (size_t)blockIdx.x * 256 + threadIdx.x;  // N*C*HW exact
    size_t r = idx & (size_t)(CH * HWD - 1);
    int c = (int)(r >> 12);
    float wg = *wgate;
    float norm = (P[idx] - meanArr[c]) * rsqrtf(varArr[c] + 1e-5f);
    out[idx] = (1.0f - wg) * x[idx] + wg * (norm * gamma[c] + beta[c]);
}

// ---------------------------------------------------------------------------
extern "C" void kernel_launch(void* const* d_in, const int* in_sizes, int n_in,
                              void* d_out, int out_size, void* d_ws, size_t ws_size,
                              hipStream_t stream)
{
    const float* x       = (const float*)d_in[0];
    const float* w_theta = (const float*)d_in[1];
    const float* b_theta = (const float*)d_in[2];
    const float* w_phi   = (const float*)d_in[3];
    const float* b_phi   = (const float*)d_in[4];
    const float* w_g     = (const float*)d_in[5];
    const float* b_g     = (const float*)d_in[6];
    const float* w_out   = (const float*)d_in[7];
    const float* b_out   = (const float*)d_in[8];
    const float* gamma   = (const float*)d_in[9];
    const float* beta    = (const float*)d_in[10];
    const float* w_gate  = (const float*)d_in[11];
    float* out = (float*)d_out;

    char* ws = (char*)d_ws;
    size_t off = 0;
    auto alloc = [&](size_t bytes) -> char* {
        char* p = ws + off;
        off += (bytes + 255) & ~(size_t)255;
        return p;
    };
    const size_t nhwc = (size_t)NB * HWD * CH;
    u16* xT  = (u16*)alloc(nhwc * 2);
    u16* wq  = (u16*)alloc((size_t)CH * CH * 2);
    u16* wk  = (u16*)alloc((size_t)CH * CH * 2);
    u16* wv  = (u16*)alloc((size_t)CH * CH * 2);
    u16* wo  = (u16*)alloc((size_t)CH * CH * 2);
    u16* Q   = (u16*)alloc(nhwc * 2);
    u16* K   = (u16*)alloc(nhwc * 2);
    u16* Vt  = (u16*)alloc(nhwc * 2);
    u16* TPG = (u16*)alloc(nhwc * 2);
    float* P    = (float*)alloc(nhwc * 4);
    float* mArr = (float*)alloc(CH * 4);
    float* vArr = (float*)alloc(CH * 4);

    convert_w_kernel<<<256, 256, 0, stream>>>(w_theta, w_phi, w_g, w_out, wq, wk, wv, wo);
    convert_x_kernel<<<dim3(HWD / 32, CH / 32, NB), dim3(32, 8), 0, stream>>>(x, xT);
    qk_kernel<<<NB * (HWD / 16), 128, 0, stream>>>(xT, wq, wk, b_theta, b_phi, Q, K);
    v_kernel<<<NB * (HWD / 16), 128, 0, stream>>>(xT, wv, b_g, Vt);
    attn_kernel<<<dim3(NB, HWD / 64), 128, 0, stream>>>(Q, K, Vt, TPG);
    pconv_kernel<<<NB * (HWD / 16), 128, 0, stream>>>(TPG, wo, b_out, P);
    bn_stats_kernel<<<CH, 256, 0, stream>>>(P, mArr, vArr);
    bn_final_kernel<<<(int)(nhwc / 256), 256, 0, stream>>>(P, x, gamma, beta, w_gate, mArr, vArr, out);
}
